// BeliefField_25228637896726
// MI455X (gfx1250) — compile-verified, matrix operand order checked by reading, not run on hardware
//
#include <hip/hip_runtime.h>
#include <math.h>

// ---------------- problem constants ----------------
static constexpr int Bb   = 64;         // batch
static constexpr int Cc   = 2048;       // channels
static constexpr int Nn   = 256;        // HW*HW spatial sites
static constexpr int Mbig = Bb * Nn;    // 16384 flattened rows

// ---------------- WMMA types ----------------
typedef __attribute__((ext_vector_type(16))) __bf16        v16bf;
typedef __attribute__((ext_vector_type(2)))  __bf16        v2bf;
typedef __attribute__((ext_vector_type(8)))  float         v8f;
typedef __attribute__((ext_vector_type(8)))  unsigned int  v8u;
typedef int v4i_vs __attribute__((vector_size(16)));       // matches async-LDS builtin param

enum { ACT_NONE = 0, ACT_SIGMOID = 1, ACT_TANH = 2, ACT_GELU = 3 };

// ---------------- CDNA5 feature probes (compile-safe fallbacks) ----------------
#if __has_builtin(__builtin_amdgcn_cvt_pk_bf16_f32)
#define HAVE_PK_BF16 1
#else
#define HAVE_PK_BF16 0
#endif

#if __has_builtin(__builtin_amdgcn_global_load_async_to_lds_b128) && \
    __has_builtin(__builtin_amdgcn_s_wait_asynccnt)
#define HAVE_ASYNC_LDS 1
#else
#define HAVE_ASYNC_LDS 0
#endif

__device__ __forceinline__ unsigned short f2bf(float x) {
  unsigned int u = __float_as_uint(x);
  return (unsigned short)((u + 0x7FFFu + ((u >> 16) & 1u)) >> 16);
}
__device__ __forceinline__ float bf2f(unsigned short h) {
  return __uint_as_float(((unsigned int)h) << 16);
}
// pack two f32 -> bf16x2 in one uint (v_cvt_pk_bf16_f32 when available)
__device__ __forceinline__ unsigned int packbf(float lo, float hi) {
#if HAVE_PK_BF16
  v2bf r = __builtin_amdgcn_cvt_pk_bf16_f32(lo, hi);
  union { v2bf b; unsigned int u; } c; c.b = r; return c.u;
#else
  return ((unsigned int)f2bf(hi) << 16) | f2bf(lo);
#endif
}
__device__ __forceinline__ float sigm(float x) { return 1.f / (1.f + expf(-x)); }
__device__ __forceinline__ float gelu_exact(float x) {
  return 0.5f * x * (1.f + erff(x * 0.70710678118654752f));
}
__device__ __forceinline__ float act_apply(float v, int act) {
  if (act == ACT_SIGMOID) return sigm(v);
  if (act == ACT_TANH)    return tanhf(v);
  if (act == ACT_GELU)    return gelu_exact(v);
  return v;
}

// ---------------- generic WMMA GEMM ----------------
// C[M,N] = act( A[M,K] @ B[K,N] + bias ) (+ optional accumulate into C)
// Block tile 128x128x32, 256 threads = 8 waves, wave tile 64x32 (4x2 WMMA).
#define BM 128
#define BN 128
#define BK 32
#define AS_STRIDE 20   // uints per LDS row (16 used + pad)
#define BS_STRIDE 20

// Load a 16x32 bf16 fragment from LDS laid out as rows of bf16 pairs.
// ISA wave32 layout: lanes 0-15 -> K 0..7 (V0-3) and 16..23 (V4-7);
// lanes 16-31 -> K 8..15 and 24..31.
__device__ __forceinline__ v16bf frag_from_lds(const unsigned int* base, int row0,
                                               int stride, int lane) {
  const unsigned int* p = base + (size_t)(row0 + (lane & 15)) * stride + ((lane >> 4) & 1) * 4;
  v8u r;
  r[0] = p[0];  r[1] = p[1];  r[2] = p[2];  r[3] = p[3];
  r[4] = p[8];  r[5] = p[9];  r[6] = p[10]; r[7] = p[11];
  union { v8u u; v16bf b; } c; c.u = r; return c.b;
}

// AMODE: 0 = A f32 row-major, 1 = A bf16 row-major, 2 = A[m,k] = src[k*lda+m] (f32)
// OUTBF: 0 = f32 out, 1 = bf16 out
// FULLM: 1 = M guaranteed multiple of BM (no row guards, enables async staging)
template <int AMODE, int OUTBF, int FULLM>
__global__ __launch_bounds__(256) void gemm_wmma(
    const void* __restrict__ Asrc, const float* __restrict__ Bsrc,
    const float* __restrict__ bias, void* __restrict__ Cdst,
    int M, int N, int K, int lda, int ldb, int ldc, int act, int accum) {
  __shared__ unsigned int As[BM * AS_STRIDE];
  __shared__ unsigned int Bs[BN * BS_STRIDE];
  const int tid  = threadIdx.x;
  const int lane = tid & 31;
  const int w    = tid >> 5;
  const int wr   = w >> 2;   // 0..1
  const int wc   = w & 3;    // 0..3
  const int m0   = blockIdx.y * BM;
  const int n0   = blockIdx.x * BN;

  v8f acc[4][2];
#pragma unroll
  for (int i = 0; i < 4; ++i)
#pragma unroll
    for (int j = 0; j < 2; ++j) { v8f z = {0, 0, 0, 0, 0, 0, 0, 0}; acc[i][j] = z; }

  for (int k0 = 0; k0 < K; k0 += BK) {
    // -------- stage A tile (bf16 pairs in LDS) --------
    if (AMODE == 0) {
      const float* A = (const float*)Asrc;
#pragma unroll
      for (int i = 0; i < 4; ++i) {
        int idx = tid + i * 256;           // 0..1023 float4 slots
        int row = idx >> 3, v4 = idx & 7;
        int gr  = m0 + row;
        float4 f = make_float4(0.f, 0.f, 0.f, 0.f);
        if (FULLM || gr < M) f = *(const float4*)(A + (size_t)gr * lda + k0 + v4 * 4);
        As[row * AS_STRIDE + v4 * 2]     = packbf(f.x, f.y);
        As[row * AS_STRIDE + v4 * 2 + 1] = packbf(f.z, f.w);
      }
    } else if (AMODE == 1) {
      const unsigned short* A = (const unsigned short*)Asrc;
#if HAVE_ASYNC_LDS
      if (FULLM) {
        // CDNA5 async Global->LDS: bf16 tile is already in the LDS byte layout.
        // 128 rows x 64B = 512 x 16B segments, 2 per thread (ASYNCcnt tracked).
#pragma unroll
        for (int i = 0; i < 2; ++i) {
          int idx = tid + i * 256;         // 0..511
          int row = idx >> 2, seg = idx & 3;
          __builtin_amdgcn_global_load_async_to_lds_b128(
              (__attribute__((address_space(1))) v4i_vs*)(A + (size_t)(m0 + row) * lda + k0 + seg * 8),
              (__attribute__((address_space(3))) v4i_vs*)&As[row * AS_STRIDE + seg * 4],
              0, 0);
        }
        __builtin_amdgcn_s_wait_asynccnt(0);
      } else
#endif
      {
#pragma unroll
        for (int i = 0; i < 8; ++i) {
          int idx = tid + i * 256;         // 0..2047 uint slots
          int row = idx >> 4, v = idx & 15;
          int gr  = m0 + row;
          unsigned int val = 0;
          if (FULLM || gr < M) val = *(const unsigned int*)(A + (size_t)gr * lda + k0 + v * 2);
          As[row * AS_STRIDE + v] = val;
        }
      }
    } else {  // AMODE == 2 : A^T source
      const float* A = (const float*)Asrc;
      unsigned short* AsU = (unsigned short*)As;
#pragma unroll
      for (int i = 0; i < 16; ++i) {
        int idx = tid + i * 256;           // 0..4095
        int kk = idx >> 7, mm = idx & 127;
        int gm = m0 + mm;
        float f = (FULLM || gm < M) ? A[(size_t)(k0 + kk) * lda + gm] : 0.f;
        AsU[mm * (AS_STRIDE * 2) + kk] = f2bf(f);
      }
    }
    // -------- stage B tile transposed: Bs[n][k] --------
    {
      unsigned short* BsU = (unsigned short*)Bs;
#pragma unroll
      for (int i = 0; i < 16; ++i) {
        int idx = tid + i * 256;
        int kk = idx >> 7, nn = idx & 127;
        float f = Bsrc[(size_t)(k0 + kk) * ldb + (n0 + nn)];
        BsU[nn * (BS_STRIDE * 2) + kk] = f2bf(f);
      }
    }
    if (k0 + BK < K)  // warm next weight tile (global_prefetch_b8)
      __builtin_prefetch((const char*)(Bsrc + (size_t)(k0 + BK) * ldb + n0 + tid), 0, 3);
    __syncthreads();

    v16bf af[4], bfr[2];
#pragma unroll
    for (int mt = 0; mt < 4; ++mt) af[mt]  = frag_from_lds(As, wr * 64 + mt * 16, AS_STRIDE, lane);
#pragma unroll
    for (int nt = 0; nt < 2; ++nt) bfr[nt] = frag_from_lds(Bs, wc * 32 + nt * 16, BS_STRIDE, lane);
#pragma unroll
    for (int mt = 0; mt < 4; ++mt)
#pragma unroll
      for (int nt = 0; nt < 2; ++nt)
        acc[mt][nt] = __builtin_amdgcn_wmma_f32_16x16x32_bf16(
            false, af[mt], false, bfr[nt], (short)0, acc[mt][nt], false, false);
    __syncthreads();
  }

  // -------- epilogue (C layout: VGPR r -> row r / r+8, lane%16 -> col) --------
  const int hi = (lane >> 4) & 1;
#pragma unroll
  for (int mt = 0; mt < 4; ++mt) {
#pragma unroll
    for (int nt = 0; nt < 2; ++nt) {
      int col = n0 + wc * 32 + nt * 16 + (lane & 15);
      float bv = bias ? bias[col] : 0.f;
#pragma unroll
      for (int r = 0; r < 8; ++r) {
        int row = m0 + wr * 64 + mt * 16 + hi * 8 + r;
        if (FULLM || row < M) {
          float v = acc[mt][nt][r] + bv;
          if (OUTBF) {
            ((unsigned short*)Cdst)[(size_t)row * ldc + col] = f2bf(act_apply(v, act));
          } else {
            float* Cp = (float*)Cdst;
            if (accum) v += Cp[(size_t)row * ldc + col];
            Cp[(size_t)row * ldc + col] = act_apply(v, act);
          }
        }
      }
    }
  }
}

// ---------------- small kernels ----------------

// coeff[i][c] = 0.25 * sigmoid(diff_alpha[i][c])
__global__ void k_coeff(const float* __restrict__ da, float* __restrict__ coeff) {
  int i = blockIdx.x * 256 + threadIdx.x;
  if (i < 2 * Cc) coeff[i] = 0.25f * sigm(da[i]);
}

// gv = sigmoid(gate_pre) * tanh(value_pre)
__global__ void k_gv(const float* __restrict__ g, const float* __restrict__ v,
                     float* __restrict__ gv) {
  int i = blockIdx.x * 256 + threadIdx.x;
  gv[i] = sigm(g[i]) * tanhf(v[i]);
}

// u_flat[b,p,c] = decay*state[b,c,p] + gv[b,c]   (tiled transpose)
__global__ __launch_bounds__(256) void k_write(const float* __restrict__ state,
                                               const float* __restrict__ gv,
                                               const float* __restrict__ log_decay,
                                               float* __restrict__ u) {
  __shared__ float tile[32][33];
  const int b = blockIdx.z, c0 = blockIdx.y * 32, p0 = blockIdx.x * 32;
  const int tx = threadIdx.x & 31, ty = threadIdx.x >> 5;
  float dec = expf(log_decay[0]);
  dec = fminf(fmaxf(dec, 0.5f), 0.99f);
#pragma unroll
  for (int it = 0; it < 4; ++it) {
    int i = ty + it * 8;  // channel offset
    tile[i][tx] = state[((size_t)b * Cc + c0 + i) * Nn + p0 + tx];
  }
  __syncthreads();
#pragma unroll
  for (int it = 0; it < 4; ++it) {
    int i = ty + it * 8;  // spatial offset
    int c = c0 + tx, p = p0 + i;
    u[((size_t)b * Nn + p) * Cc + c] = dec * tile[tx][i] + gv[(size_t)b * Cc + c];
  }
}

// out[b,c,p] = u_flat[b,p,c]   (inverse transpose for BCHW output)
__global__ __launch_bounds__(256) void k_readout(const float* __restrict__ u,
                                                 float* __restrict__ out) {
  __shared__ float tile[32][33];
  const int b = blockIdx.z, c0 = blockIdx.y * 32, p0 = blockIdx.x * 32;
  const int tx = threadIdx.x & 31, ty = threadIdx.x >> 5;
#pragma unroll
  for (int it = 0; it < 4; ++it) {
    int i = ty + it * 8;  // spatial offset
    tile[i][tx] = u[((size_t)b * Nn + p0 + i) * Cc + c0 + tx];
  }
  __syncthreads();
#pragma unroll
  for (int it = 0; it < 4; ++it) {
    int i = ty + it * 8;  // channel offset
    out[((size_t)b * Cc + c0 + i) * Nn + p0 + tx] = tile[tx][i];
  }
}

// hid[b,c] = gelu(stimulus[b,:] @ s1_W + s1_b)   (K = 6, tiny)
__global__ void k_stim1(const float* __restrict__ stimulus, const float* __restrict__ s1_W,
                        const float* __restrict__ s1_b, float* __restrict__ hid) {
  int i = blockIdx.x * 256 + threadIdx.x;   // < B*C
  int b = i >> 11, c = i & (Cc - 1);
  float s = s1_b[c];
#pragma unroll
  for (int k = 0; k < 6; ++k) s += stimulus[b * 6 + k] * s1_W[k * Cc + c];
  hid[i] = gelu_exact(s);
}

// diff[b,p,c] = sum_i coeff[i][c] * lap(u, d_i)   -> bf16
__global__ __launch_bounds__(256) void k_lap(const float* __restrict__ u,
                                             const float* __restrict__ coeff,
                                             unsigned short* __restrict__ diff) {
  const int bp = blockIdx.x;               // b*256 + p
  const int p = bp & 255;
  const int y = p >> 4, x = p & 15;
  const size_t ro = (size_t)bp * Cc;
  // dilation 1
  const float* u1u = (y >= 1)  ? u + ro - 16 * Cc : nullptr;
  const float* u1d = (y <= 14) ? u + ro + 16 * Cc : nullptr;
  const float* u1l = (x >= 1)  ? u + ro - Cc      : nullptr;
  const float* u1r = (x <= 14) ? u + ro + Cc      : nullptr;
  // dilation 4
  const float* u4u = (y >= 4)  ? u + ro - 64 * Cc : nullptr;
  const float* u4d = (y <= 11) ? u + ro + 64 * Cc : nullptr;
  const float* u4l = (x >= 4)  ? u + ro - 4 * Cc  : nullptr;
  const float* u4r = (x <= 11) ? u + ro + 4 * Cc  : nullptr;
  for (int c = threadIdx.x; c < Cc; c += 256) {
    float ctr = u[ro + c];
    float s1 = -4.f * ctr;
    if (u1u) s1 += u1u[c];
    if (u1d) s1 += u1d[c];
    if (u1l) s1 += u1l[c];
    if (u1r) s1 += u1r[c];
    float s4 = -4.f * ctr;
    if (u4u) s4 += u4u[c];
    if (u4d) s4 += u4d[c];
    if (u4l) s4 += u4l[c];
    if (u4r) s4 += u4r[c];
    diff[ro + c] = f2bf(coeff[c] * s1 + coeff[Cc + c] * s4);
  }
}

// pooled[b,c] = mean_p u[b,p,c]
__global__ __launch_bounds__(256) void k_pool(const float* __restrict__ u,
                                              float* __restrict__ pooled) {
  int b = blockIdx.x >> 3;
  int c = ((blockIdx.x & 7) << 8) + threadIdx.x;
  const float* p = u + (size_t)b * Nn * Cc + c;
  float s = 0.f;
  for (int i = 0; i < Nn; ++i) s += p[(size_t)i * Cc];
  pooled[(size_t)b * Cc + c] = s * (1.f / (float)Nn);
}

// Wh = 0.99*Wh + (0.01/16384)*corr_raw
__global__ void k_heb(float* __restrict__ Wh, const float* __restrict__ corr) {
  size_t i = (size_t)blockIdx.x * 256 + threadIdx.x;
  Wh[i] = 0.99f * Wh[i] + (0.01f / (float)Mbig) * corr[i];
}

// GRU combine: h = (1-z)*tanh(in + r*hn) + z*h
__global__ void k_gru(const float* __restrict__ gi, const float* __restrict__ gh,
                      float* __restrict__ h) {
  int i = blockIdx.x * 256 + threadIdx.x;   // < B*C
  int b = i >> 11, c = i & (Cc - 1);
  const float* gib = gi + (size_t)b * 3 * Cc;
  const float* ghb = gh + (size_t)b * 3 * Cc;
  float r  = sigm(gib[c] + ghb[c]);
  float z  = sigm(gib[Cc + c] + ghb[Cc + c]);
  float nc = tanhf(gib[2 * Cc + c] + r * ghb[2 * Cc + c]);
  float hp = h[i];
  h[i] = (1.f - z) * nc + z * hp;
}

// u += dt*(du + diff + stim + ap*h) ; optional RMS-norm over channels
__global__ __launch_bounds__(256) void k_update(float* __restrict__ u,
                                                const float* __restrict__ du,
                                                const unsigned short* __restrict__ diff,
                                                const float* __restrict__ stim,
                                                const float* __restrict__ h,
                                                const float* __restrict__ rms_w,
                                                const float* __restrict__ alpha_pump,
                                                const float* __restrict__ log_dt,
                                                int do_rms) {
  __shared__ float red[256];
  const int bp = blockIdx.x;
  const int b  = bp >> 8;
  const size_t ro = (size_t)bp * Cc;
  float dt = expf(log_dt[0]);
  dt = fminf(fmaxf(dt, 0.01f), 0.3f);
  float apx = alpha_pump[0];
  float ap  = (apx > 20.f) ? apx : log1pf(expf(apx));
  float vals[8];
  float ss = 0.f;
#pragma unroll
  for (int j = 0; j < 8; ++j) {
    int c = threadIdx.x + j * 256;
    float xv = u[ro + c] + dt * (du[ro + c] + bf2f(diff[ro + c]) +
                                 stim[(size_t)b * Cc + c] + ap * h[(size_t)b * Cc + c]);
    vals[j] = xv;
    ss += xv * xv;
  }
  if (do_rms) {
    red[threadIdx.x] = ss;
    __syncthreads();
    for (int s = 128; s > 0; s >>= 1) {
      if (threadIdx.x < s) red[threadIdx.x] += red[threadIdx.x + s];
      __syncthreads();
    }
    float inv = rsqrtf(red[0] * (1.f / (float)Cc) + 1e-6f);
#pragma unroll
    for (int j = 0; j < 8; ++j) {
      int c = threadIdx.x + j * 256;
      vals[j] *= inv * rms_w[c];
    }
  }
#pragma unroll
  for (int j = 0; j < 8; ++j) u[ro + threadIdx.x + j * 256] = vals[j];
}

// LayerNorm over channels of pooled -> pnorm
__global__ __launch_bounds__(256) void k_ln(const float* __restrict__ pooled,
                                            const float* __restrict__ g,
                                            const float* __restrict__ beta,
                                            float* __restrict__ out) {
  __shared__ float red[256];
  const int b = blockIdx.x;
  const float* p = pooled + (size_t)b * Cc;
  float v[8];
  float s = 0.f;
#pragma unroll
  for (int j = 0; j < 8; ++j) { v[j] = p[threadIdx.x + j * 256]; s += v[j]; }
  red[threadIdx.x] = s;
  __syncthreads();
  for (int st = 128; st > 0; st >>= 1) {
    if (threadIdx.x < st) red[threadIdx.x] += red[threadIdx.x + st];
    __syncthreads();
  }
  float mu = red[0] * (1.f / (float)Cc);
  __syncthreads();
  float ss = 0.f;
#pragma unroll
  for (int j = 0; j < 8; ++j) { float d = v[j] - mu; ss += d * d; }
  red[threadIdx.x] = ss;
  __syncthreads();
  for (int st = 128; st > 0; st >>= 1) {
    if (threadIdx.x < st) red[threadIdx.x] += red[threadIdx.x + st];
    __syncthreads();
  }
  float inv = rsqrtf(red[0] * (1.f / (float)Cc) + 1e-5f);
#pragma unroll
  for (int j = 0; j < 8; ++j) {
    int c = threadIdx.x + j * 256;
    out[(size_t)b * Cc + c] = (v[j] - mu) * inv * g[c] + beta[c];
  }
}

// ---------------- workspace layout (bytes) ----------------
static constexpr size_t OFF_U     = 0;                                   // f32  16384x2048
static constexpr size_t OFF_DU    = OFF_U    + (size_t)Mbig * Cc * 4;    // f32  16384x2048
static constexpr size_t OFF_DIFF  = OFF_DU   + (size_t)Mbig * Cc * 4;    // bf16 16384x2048
static constexpr size_t OFF_MID   = OFF_DIFF + (size_t)Mbig * Cc * 2;    // bf16 16384x2048
static constexpr size_t OFF_WH    = OFF_MID  + (size_t)Mbig * Cc * 2;    // f32  2048x2048
static constexpr size_t OFF_CORR  = OFF_WH   + (size_t)Cc * Cc * 4;      // f32  2048x2048
static constexpr size_t OFF_H     = OFF_CORR + (size_t)Cc * Cc * 4;      // f32  64x2048
static constexpr size_t OFF_POOL  = OFF_H    + (size_t)Bb * Cc * 4;      // f32  64x2048
static constexpr size_t OFF_GI    = OFF_POOL + (size_t)Bb * Cc * 4;      // f32  64x6144
static constexpr size_t OFF_GH    = OFF_GI   + (size_t)Bb * 3 * Cc * 4;  // f32  64x6144
static constexpr size_t OFF_STIM  = OFF_GH   + (size_t)Bb * 3 * Cc * 4;  // f32  64x2048
static constexpr size_t OFF_HID   = OFF_STIM + (size_t)Bb * Cc * 4;      // f32  64x2048
static constexpr size_t OFF_GV    = OFF_HID  + (size_t)Bb * Cc * 4;      // f32  64x2048
static constexpr size_t OFF_PN    = OFF_GV   + (size_t)Bb * Cc * 4;      // f32  64x2048
static constexpr size_t OFF_COEFF = OFF_PN   + (size_t)Bb * Cc * 4;      // f32  2x2048

extern "C" void kernel_launch(void* const* d_in, const int* in_sizes, int n_in,
                              void* d_out, int out_size, void* d_ws, size_t ws_size,
                              hipStream_t stream) {
  (void)in_sizes; (void)n_in; (void)out_size; (void)ws_size;
  const float* state       = (const float*)d_in[0];
  const float* observation = (const float*)d_in[1];
  const float* stimulus    = (const float*)d_in[2];
  const float* h_global    = (const float*)d_in[3];
  const float* heb_W       = (const float*)d_in[4];
  const float* gate_W      = (const float*)d_in[5];
  const float* gate_b      = (const float*)d_in[6];
  const float* value_W     = (const float*)d_in[7];
  const float* value_b     = (const float*)d_in[8];
  const float* diff_alpha  = (const float*)d_in[9];
  const float* r1_W        = (const float*)d_in[10];
  const float* r1_b        = (const float*)d_in[11];
  const float* r2_W        = (const float*)d_in[12];
  const float* r2_b        = (const float*)d_in[13];
  const float* rms_w       = (const float*)d_in[14];
  const float* gru_Wih     = (const float*)d_in[15];
  const float* gru_Whh     = (const float*)d_in[16];
  const float* gru_bih     = (const float*)d_in[17];
  const float* gru_bhh     = (const float*)d_in[18];
  const float* alpha_pump  = (const float*)d_in[19];
  const float* s1_W        = (const float*)d_in[20];
  const float* s1_b        = (const float*)d_in[21];
  const float* s2_W        = (const float*)d_in[22];
  const float* s2_b        = (const float*)d_in[23];
  const float* ln_g        = (const float*)d_in[24];
  const float* ln_b        = (const float*)d_in[25];
  const float* read_W      = (const float*)d_in[26];
  const float* read_b      = (const float*)d_in[27];
  const float* log_decay   = (const float*)d_in[28];
  const float* log_dt      = (const float*)d_in[29];

  char* ws = (char*)d_ws;
  float*          u_flat = (float*)(ws + OFF_U);
  float*          du     = (float*)(ws + OFF_DU);
  unsigned short* diff   = (unsigned short*)(ws + OFF_DIFF);
  unsigned short* mid    = (unsigned short*)(ws + OFF_MID);
  float*          Wh     = (float*)(ws + OFF_WH);
  float*          corr   = (float*)(ws + OFF_CORR);
  float*          h      = (float*)(ws + OFF_H);
  float*          pooled = (float*)(ws + OFF_POOL);
  float*          gi     = (float*)(ws + OFF_GI);
  float*          gh     = (float*)(ws + OFF_GH);
  float*          stim   = (float*)(ws + OFF_STIM);
  float*          hid    = (float*)(ws + OFF_HID);
  float*          gv     = (float*)(ws + OFF_GV);
  float*          pnorm  = (float*)(ws + OFF_PN);
  float*          coeff  = (float*)(ws + OFF_COEFF);

  float* out_u   = (float*)d_out;
  float* out_rep = out_u + (size_t)Bb * Cc * Nn;

  // ---- init persistent traces ----
  (void)hipMemcpyAsync(Wh, heb_W, (size_t)Cc * Cc * 4, hipMemcpyDeviceToDevice, stream);
  (void)hipMemcpyAsync(h, h_global, (size_t)Bb * Cc * 4, hipMemcpyDeviceToDevice, stream);
  k_coeff<<<(2 * Cc + 255) / 256, 256, 0, stream>>>(diff_alpha, coeff);

  const dim3 gSmall(Cc / BN, (Bb + BM - 1) / BM);       // 16 x 1
  const dim3 gGru(3 * Cc / BN, (Bb + BM - 1) / BM);     // 48 x 1
  const dim3 gBig(Cc / BN, Mbig / BM);                  // 16 x 128
  const dim3 gCorr(Cc / BN, Cc / BM);                   // 16 x 16
  const dim3 gTr(Nn / 32, Cc / 32, Bb);                 // 8 x 64 x 64

  // ---- write phase: gate/value GEMMs + field init ----
  gemm_wmma<0, 0, 0><<<gSmall, 256, 0, stream>>>(observation, gate_W, gate_b, gi,
                                                 Bb, Cc, Cc, Cc, Cc, Cc, ACT_NONE, 0);
  gemm_wmma<0, 0, 0><<<gSmall, 256, 0, stream>>>(observation, value_W, value_b, gh,
                                                 Bb, Cc, Cc, Cc, Cc, Cc, ACT_NONE, 0);
  k_gv<<<(Bb * Cc) / 256, 256, 0, stream>>>(gi, gh, gv);
  k_write<<<gTr, 256, 0, stream>>>(state, gv, log_decay, u_flat);

  // ---- stimulus MLP (computed once) ----
  k_stim1<<<(Bb * Cc) / 256, 256, 0, stream>>>(stimulus, s1_W, s1_b, hid);
  gemm_wmma<0, 0, 0><<<gSmall, 256, 0, stream>>>(hid, s2_W, s2_b, stim,
                                                 Bb, Cc, Cc, Cc, Cc, Cc, ACT_NONE, 0);

  // ---- evolution steps ----
  for (int step = 0; step < 3; ++step) {
    k_lap<<<Mbig, 256, 0, stream>>>(u_flat, coeff, diff);
    k_pool<<<Bb * (Cc / 256), 256, 0, stream>>>(u_flat, pooled);

    // Hebbian trace: corr = u^T u (raw), Wh = 0.99 Wh + (0.01/16384) corr
    gemm_wmma<2, 0, 1><<<gCorr, 256, 0, stream>>>(u_flat, u_flat, nullptr, corr,
                                                  Cc, Cc, Mbig, Cc, Cc, Cc, ACT_NONE, 0);
    k_heb<<<(Cc * Cc) / 256, 256, 0, stream>>>(Wh, corr);

    // GRU on pooled state
    gemm_wmma<0, 0, 0><<<gGru, 256, 0, stream>>>(pooled, gru_Wih, gru_bih, gi,
                                                 Bb, 3 * Cc, Cc, Cc, 3 * Cc, 3 * Cc, ACT_NONE, 0);
    gemm_wmma<0, 0, 0><<<gGru, 256, 0, stream>>>(h, gru_Whh, gru_bhh, gh,
                                                 Bb, 3 * Cc, Cc, Cc, 3 * Cc, 3 * Cc, ACT_NONE, 0);
    k_gru<<<(Bb * Cc) / 256, 256, 0, stream>>>(gi, gh, h);

    // reaction: mid = gelu(u @ r1 + b) (bf16), du = mid @ r2 + b, du += diff @ Wh
    gemm_wmma<0, 1, 1><<<gBig, 256, 0, stream>>>(u_flat, r1_W, r1_b, mid,
                                                 Mbig, Cc, Cc, Cc, Cc, Cc, ACT_GELU, 0);
    gemm_wmma<1, 0, 1><<<gBig, 256, 0, stream>>>(mid, r2_W, r2_b, du,
                                                 Mbig, Cc, Cc, Cc, Cc, Cc, ACT_NONE, 0);
    gemm_wmma<1, 0, 1><<<gBig, 256, 0, stream>>>(diff, Wh, nullptr, du,
                                                 Mbig, Cc, Cc, Cc, Cc, Cc, ACT_NONE, 1);

    k_update<<<Mbig, 256, 0, stream>>>(u_flat, du, diff, stim, h, rms_w,
                                       alpha_pump, log_dt, ((step + 1) % 2 == 0) ? 1 : 0);
  }

  // ---- read phase ----
  k_pool<<<Bb * (Cc / 256), 256, 0, stream>>>(u_flat, pooled);
  k_ln<<<Bb, 256, 0, stream>>>(pooled, ln_g, ln_b, pnorm);
  gemm_wmma<0, 0, 0><<<gSmall, 256, 0, stream>>>(pnorm, read_W, read_b, out_rep,
                                                 Bb, Cc, Cc, Cc, Cc, Cc, ACT_NONE, 0);
  k_readout<<<gTr, 256, 0, stream>>>(u_flat, out_u);
}
